// ANYCSP_55860344651814
// MI455X (gfx1250) — matrix operand order repaired and checked by reading, not compiled, not run on hardware
//
#include <hip/hip_runtime.h>
#include <hip/hip_bf16.h>
#include <math.h>

#define H     128
#define NVAR  20000
#define DDOM  8
#define NVAL  (NVAR * DDOM)   // 160000
#define NCST  20000
#define DEG   8
#define NE    (NCST * DEG)    // 160000
#define NSTEPS 4

typedef __attribute__((ext_vector_type(16))) __bf16 bf16x16;
typedef __attribute__((ext_vector_type(8)))  __bf16 bf16x8;
typedef __attribute__((ext_vector_type(8)))  float  f32x8;

// ---------------------------------------------------------------- WMMA core
__device__ __forceinline__ f32x8 wmma_bf16(bf16x16 a, bf16x16 b, f32x8 c) {
  return __builtin_amdgcn_wmma_f32_16x16x32_bf16(
      /*neg_a=*/false, a, /*neg_b=*/false, b,
      /*c_mod=*/(short)0, c, /*reuse_a=*/false, /*reuse_b=*/false);
}

// A fragment: 16x32 bf16 tile. rp = pointer to this lane's row (row-major, K-contig).
// 16-bit A layout (ISA 7.12.2): lanes 0-15 hold K=kb+0..7 & kb+16..23 with kb=0,
// lanes 16-31 same rows with kb=8.
__device__ __forceinline__ bf16x16 load_afrag(const __bf16* rp, int kc, int lane) {
  int kb = kc * 32 + ((lane >> 4) << 3);
  bf16x8 lo = *(const bf16x8*)(rp + kb);
  bf16x8 hi = *(const bf16x8*)(rp + kb + 16);
  return __builtin_shufflevector(lo, hi, 0,1,2,3,4,5,6,7,8,9,10,11,12,13,14,15);
}

// B fragment from pre-packed weights: one 32B load per lane.
__device__ __forceinline__ bf16x16 load_bfrag(const __bf16* pk, int ntiles, int kc,
                                              int nt, int lane) {
  return ((const bf16x16*)pk)[(size_t)(kc * ntiles + nt) * 32 + lane];
}

// ------------------------------------------------------------- weight pack
// Pack f32 weight W[K x N] (row-major) into B-fragment order:
// lane L: n = nt*16 + (L&15), k = kc*32 + ((e>>3)<<4) + ((L>>4)<<3) + (e&7)  (table order, matches A).
__global__ void k_pack(const float* __restrict__ W, __bf16* __restrict__ pk,
                       int K, int N) {
  int tid = blockIdx.x * blockDim.x + threadIdx.x;
  if (tid >= K * N) return;
  int e    = tid & 15;
  int lane = (tid >> 4) & 31;
  int t    = tid >> 9;
  int ntiles = N >> 4;
  int nt = t % ntiles;
  int kc = t / ntiles;
  int n = nt * 16 + (lane & 15);
  int k = kc * 32 + ((e >> 3) << 4) + ((lane >> 4) << 3) + (e & 7);
  pk[tid] = (__bf16)W[(size_t)k * N + n];
}

// ------------------------------------------------------------- elementwise
__global__ void k_f32_to_bf16(const float* __restrict__ s, __bf16* __restrict__ d,
                              size_t n) {
  size_t i = (size_t)blockIdx.x * blockDim.x + threadIdx.x;
  size_t st = (size_t)gridDim.x * blockDim.x;
  for (; i < n; i += st) d[i] = (__bf16)s[i];
}

__global__ void k_zero_f32(float* __restrict__ p, size_t n) {
  size_t i = (size_t)blockIdx.x * blockDim.x + threadIdx.x;
  size_t st = (size_t)gridDim.x * blockDim.x;
  for (; i < n; i += st) p[i] = 0.f;
}

__global__ void k_h0(const float* __restrict__ hinit, float* __restrict__ h_val,
                     __bf16* __restrict__ hv) {
  size_t n = (size_t)NVAL * H;
  size_t i = (size_t)blockIdx.x * blockDim.x + threadIdx.x;
  size_t st = (size_t)gridDim.x * blockDim.x;
  for (; i < n; i += st) {
    float v = hinit[i & (H - 1)];
    h_val[i] = v;
    hv[i] = (__bf16)v;
  }
}

__global__ void k_init(float* __restrict__ dout, float* __restrict__ sat_acc,
                       float* __restrict__ best) {
  int t = threadIdx.x;
  if (t < 9) dout[t] = 0.f;
  if (t == 0) { *sat_acc = 0.f; *best = 3.4e38f; }
}

// ------------------------------------------------- generic GEMM (direct A)
// out[rows x N] = op(A[rows x K] @ W + bias), op = relu or identity, bf16 out.
__global__ void k_gemm(const __bf16* __restrict__ A, int K,
                       const __bf16* __restrict__ pkW, int N,
                       const float* __restrict__ bias, __bf16* __restrict__ out,
                       int relu) {
  int lane = threadIdx.x & 31;
  int wv   = threadIdx.x >> 5;
  int rt   = blockIdx.x;
  int ntiles = N >> 4, kchunks = K >> 5;
  int m = lane & 15, half = lane >> 4;
  const __bf16* rp = A + (size_t)(rt * 16 + m) * K;
  for (int nt = wv; nt < ntiles; nt += 8) {
    f32x8 acc = {};
    for (int kc = 0; kc < kchunks; ++kc)
      acc = wmma_bf16(load_afrag(rp, kc, lane), load_bfrag(pkW, ntiles, kc, nt, lane), acc);
    int n = nt * 16 + m;
    float bs = bias[n];
#pragma unroll
    for (int r = 0; r < 8; ++r) {
      float v = acc[r] + bs;
      if (relu) v = fmaxf(v, 0.f);
      out[(size_t)(rt * 16 + r + 8 * half) * N + n] = (__bf16)v;
    }
  }
}

// ------------------------------------------------- edge MLP stage 1
// t1[e] = relu(h_val[val_idx[e]] @ W1[0:128] + a_e * W1[128,:] + b1)
__global__ void k_edge_mlp1(const __bf16* __restrict__ hv,
                            const float* __restrict__ assign,
                            const int* __restrict__ val_idx,
                            const __bf16* __restrict__ pkW1,
                            const __bf16* __restrict__ w1a,
                            const float* __restrict__ b1,
                            __bf16* __restrict__ out) {
  int lane = threadIdx.x & 31;
  int wv   = threadIdx.x >> 5;     // ntile 0..7
  int et   = blockIdx.x;
  int m = lane & 15, half = lane >> 4;
  const __bf16* rp = hv + (size_t)val_idx[et * 16 + m] * H;
  f32x8 acc = {};
  for (int kc = 0; kc < 4; ++kc)
    acc = wmma_bf16(load_afrag(rp, kc, lane), load_bfrag(pkW1, 8, kc, wv, lane), acc);
  int n = wv * 16 + m;
  float wa = (float)w1a[n], bs = b1[n];
#pragma unroll
  for (int r = 0; r < 8; ++r) {
    int e = et * 16 + r + 8 * half;
    float ae = assign[val_idx[e]];
    float v = acc[r] + ae * wa + bs;
    out[(size_t)e * H + n] = (__bf16)fmaxf(v, 0.f);
  }
}

// ------------------------------------------------- edge MLP stage 2 + segsum
// m = t1 @ W2; r_cst[c] = sum over the 8 contiguous edges of constraint c.
__global__ void k_edge_mlp2(const __bf16* __restrict__ t1,
                            const __bf16* __restrict__ pkW2,
                            float* __restrict__ r_cst) {
  int lane = threadIdx.x & 31;
  int wv   = threadIdx.x >> 5;
  int et   = blockIdx.x;
  int m = lane & 15, half = lane >> 4;
  const __bf16* rp = t1 + (size_t)(et * 16 + m) * H;
  f32x8 acc = {};
  for (int kc = 0; kc < 4; ++kc)
    acc = wmma_bf16(load_afrag(rp, kc, lane), load_bfrag(pkW2, 8, kc, wv, lane), acc);
  float s = 0.f;
#pragma unroll
  for (int r = 0; r < 8; ++r) s += acc[r];
  int c = et * 2 + half;           // cst_idx = repeat(arange, 8)
  r_cst[(size_t)c * H + wv * 16 + m] = s;
}

// ------------------------------------------------- cst2val edge + scatter
// y_e = relu([r_cst[e/8], x_val[val_idx[e]]] @ Wc + bc); atomic scatter to y_val.
__global__ void k_cst2val(const __bf16* __restrict__ rc,
                          const __bf16* __restrict__ xv,
                          const int* __restrict__ val_idx,
                          const __bf16* __restrict__ pkWc,
                          const float* __restrict__ bc,
                          float* __restrict__ y_val) {
  int lane = threadIdx.x & 31;
  int wv   = threadIdx.x >> 5;
  int et   = blockIdx.x;
  int m = lane & 15, half = lane >> 4;
  int e_row = et * 16 + m;
  const __bf16* rpA = rc + (size_t)(e_row >> 3) * H;
  const __bf16* rpB = xv + (size_t)val_idx[e_row] * H;
  f32x8 acc = {};
  for (int kc = 0; kc < 8; ++kc) {
    bf16x16 a = (kc < 4) ? load_afrag(rpA, kc, lane) : load_afrag(rpB, kc - 4, lane);
    acc = wmma_bf16(a, load_bfrag(pkWc, 8, kc, wv, lane), acc);
  }
  int n = wv * 16 + m;
  float bs = bc[n];
#pragma unroll
  for (int r = 0; r < 8; ++r) {
    int e = et * 16 + r + 8 * half;
    float v = fmaxf(acc[r] + bs, 0.f);
    atomicAdd(&y_val[(size_t)val_idx[e] * H + n], v);
  }
}

// ------------------------------------------------- domain mean
__global__ void k_ymean(const float* __restrict__ y_val, __bf16* __restrict__ ym) {
  int tid = blockIdx.x * blockDim.x + threadIdx.x;
  if (tid >= NVAR * H) return;
  int var = tid >> 7, h = tid & 127;
  float s = 0.f;
#pragma unroll
  for (int d = 0; d < 8; ++d) s += y_val[((size_t)var * 8 + d) * H + h];
  ym[tid] = (__bf16)(s * 0.125f);
}

// ------------------------------------------------- val2val
// z = relu([y_val[row], ym[row/8]] @ Wv + bv)
__global__ void k_val2val(const __bf16* __restrict__ yv,
                          const __bf16* __restrict__ ym,
                          const __bf16* __restrict__ pkWv,
                          const float* __restrict__ bv,
                          __bf16* __restrict__ z) {
  int lane = threadIdx.x & 31;
  int wv   = threadIdx.x >> 5;
  int rt   = blockIdx.x;
  int m = lane & 15, half = lane >> 4;
  int row = rt * 16 + m;
  const __bf16* rpA = yv + (size_t)row * H;
  const __bf16* rpB = ym + (size_t)(row >> 3) * H;
  f32x8 acc = {};
  for (int kc = 0; kc < 8; ++kc) {
    bf16x16 a = (kc < 4) ? load_afrag(rpA, kc, lane) : load_afrag(rpB, kc - 4, lane);
    acc = wmma_bf16(a, load_bfrag(pkWv, 8, kc, wv, lane), acc);
  }
  int n = wv * 16 + m;
  float bs = bv[n];
#pragma unroll
  for (int r = 0; r < 8; ++r)
    z[(size_t)(rt * 16 + r + 8 * half) * H + n] = (__bf16)fmaxf(acc[r] + bs, 0.f);
}

// ------------------------------------------------- GRU pointwise
__global__ void k_gru(const __bf16* __restrict__ gi, const __bf16* __restrict__ gh,
                      float* __restrict__ h_val, __bf16* __restrict__ hv) {
  size_t n = (size_t)NVAL * H;
  size_t i = (size_t)blockIdx.x * blockDim.x + threadIdx.x;
  size_t st = (size_t)gridDim.x * blockDim.x;
  for (; i < n; i += st) {
    size_t row = i >> 7; int j = (int)(i & 127);
    size_t b3 = row * 384;
    float ir = (float)gi[b3 + j],       hr = (float)gh[b3 + j];
    float iz = (float)gi[b3 + 128 + j], hz = (float)gh[b3 + 128 + j];
    float in = (float)gi[b3 + 256 + j], hn = (float)gh[b3 + 256 + j];
    float r  = 1.f / (1.f + expf(-(ir + hr)));
    float zz = 1.f / (1.f + expf(-(iz + hz)));
    float nn = tanhf(in + r * hn);
    float h  = h_val[i];
    float o  = (1.f - zz) * nn + zz * h;
    h_val[i] = o;
    hv[i] = (__bf16)o;
  }
}

// ------------------------------------------------- policy head tail
__global__ void k_logits(const __bf16* __restrict__ t, const float* __restrict__ Wp2,
                         float* __restrict__ logits) {
  int row = blockIdx.x * blockDim.x + threadIdx.x;
  if (row >= NVAL) return;
  const __bf16* p = t + (size_t)row * H;
  float s = 0.f;
#pragma unroll 8
  for (int j = 0; j < H; ++j) s += (float)p[j] * Wp2[j];
  logits[row] = s;
}

// ------------------------------------------------- Gumbel-max sample + logp
__global__ void k_sample(const float* __restrict__ logits, const float* __restrict__ g,
                         float* __restrict__ assign, float* __restrict__ dout, int slot) {
  int v = blockIdx.x * blockDim.x + threadIdx.x;
  if (v >= NVAR) return;
  int base = v * 8;
  float l[8];
#pragma unroll
  for (int d = 0; d < 8; ++d) l[d] = logits ? logits[base + d] : 0.f;
  int c = 0; float bestv = l[0] + g[base];
#pragma unroll
  for (int d = 1; d < 8; ++d) {
    float s = l[d] + g[base + d];
    if (s > bestv) { bestv = s; c = d; }
  }
#pragma unroll
  for (int d = 0; d < 8; ++d) assign[base + d] = (d == c) ? 1.f : 0.f;
  if (slot >= 0) {
    float mx = l[0];
#pragma unroll
    for (int d = 1; d < 8; ++d) mx = fmaxf(mx, l[d]);
    float se = 0.f;
#pragma unroll
    for (int d = 0; d < 8; ++d) se += expf(l[d] - mx);
    atomicAdd(&dout[slot], l[c] - (mx + logf(se)));
  }
}

// ------------------------------------------------- unsat (segment max)
__global__ void k_unsat(const float* __restrict__ assign, const int* __restrict__ val_idx,
                        float* __restrict__ sat_acc) {
  int c = blockIdx.x * blockDim.x + threadIdx.x;
  if (c >= NCST) return;
  float sat = 0.f;
#pragma unroll
  for (int j = 0; j < DEG; ++j) sat = fmaxf(sat, assign[val_idx[c * DEG + j]]);
  atomicAdd(sat_acc, sat);
}

__global__ void k_finalize(float* __restrict__ sat_acc, float* __restrict__ best,
                           int slot, float* __restrict__ dout) {
  float nu = (float)NCST - *sat_acc;
  *sat_acc = 0.f;
  float b = fminf(*best, nu);
  *best = b;
  if (slot >= 0) dout[4 + slot] = nu;
  dout[8] = b;
}

// ================================================================= launch
static inline size_t al256(size_t x) { return (x + 255) & ~(size_t)255; }

extern "C" void kernel_launch(void* const* d_in, const int* in_sizes, int n_in,
                              void* d_out, int out_size, void* d_ws, size_t ws_size,
                              hipStream_t stream) {
  const float* h_init = (const float*)d_in[0];
  const float* W1  = (const float*)d_in[1];
  const float* b1  = (const float*)d_in[2];
  const float* W2  = (const float*)d_in[3];
  const float* Wx  = (const float*)d_in[4];
  const float* bx  = (const float*)d_in[5];
  const float* Wc  = (const float*)d_in[6];
  const float* bc  = (const float*)d_in[7];
  const float* Wv  = (const float*)d_in[8];
  const float* bv  = (const float*)d_in[9];
  const float* Wi  = (const float*)d_in[10];
  const float* Wh  = (const float*)d_in[11];
  const float* bi  = (const float*)d_in[12];
  const float* bh  = (const float*)d_in[13];
  const float* Wp1 = (const float*)d_in[14];
  const float* bp1 = (const float*)d_in[15];
  const float* Wp2 = (const float*)d_in[16];
  const float* gumbel  = (const float*)d_in[17];
  const int*   val_idx = (const int*)d_in[18];
  float* dout = (float*)d_out;

  // ---- workspace layout
  char* p = (char*)d_ws;
  auto alloc = [&](size_t bytes) -> char* { char* r = p; p += al256(bytes); return r; };
  float*  h_val = (float*) alloc((size_t)NVAL * H * 4);
  __bf16* hv    = (__bf16*)alloc((size_t)NVAL * H * 2);
  __bf16* sb    = (__bf16*)alloc((size_t)NE   * H * 2);   // t1 / z / p1out
  __bf16* xy    = (__bf16*)alloc((size_t)NVAL * H * 2);   // x_val then y_val bf16
  float*  r_cst = (float*) alloc((size_t)NCST * H * 4);
  __bf16* rc    = (__bf16*)alloc((size_t)NCST * H * 2);
  float*  y_val = (float*) alloc((size_t)NVAL * H * 4);
  __bf16* ym    = (__bf16*)alloc((size_t)NVAR * H * 2);
  __bf16* gi    = (__bf16*)alloc((size_t)NVAL * 384 * 2);
  __bf16* gh    = (__bf16*)alloc((size_t)NVAL * 384 * 2);
  float*  logits= (float*) alloc((size_t)NVAL * 4);
  float*  assign= (float*) alloc((size_t)NVAL * 4);
  __bf16* W1p   = (__bf16*)alloc(128 * 128 * 2);
  __bf16* w1a   = (__bf16*)alloc(128 * 2);
  __bf16* W2p   = (__bf16*)alloc(128 * 128 * 2);
  __bf16* Wxp   = (__bf16*)alloc(128 * 128 * 2);
  __bf16* Wcp   = (__bf16*)alloc(256 * 128 * 2);
  __bf16* Wvp   = (__bf16*)alloc(256 * 128 * 2);
  __bf16* Wip   = (__bf16*)alloc(128 * 384 * 2);
  __bf16* Whp   = (__bf16*)alloc(128 * 384 * 2);
  __bf16* Wp1p  = (__bf16*)alloc(128 * 128 * 2);
  float*  sat_acc = (float*)alloc(4);
  float*  best    = (float*)alloc(4);

  const int T = 256;
  const int ET = NE / 16;     // 10000 edge tiles
  const int VT = NVAL / 16;   // 10000 value-row tiles
  dim3 b256(T);

  k_init<<<1, 32, 0, stream>>>(dout, sat_acc, best);

  // pack weights (per-launch; tiny)
  k_pack<<<(128 * 128 + T - 1) / T, b256, 0, stream>>>(W1, W1p, 128, 128);
  k_f32_to_bf16<<<1, 128, 0, stream>>>(W1 + 128 * 128, w1a, 128);
  k_pack<<<(128 * 128 + T - 1) / T, b256, 0, stream>>>(W2, W2p, 128, 128);
  k_pack<<<(128 * 128 + T - 1) / T, b256, 0, stream>>>(Wx, Wxp, 128, 128);
  k_pack<<<(256 * 128 + T - 1) / T, b256, 0, stream>>>(Wc, Wcp, 256, 128);
  k_pack<<<(256 * 128 + T - 1) / T, b256, 0, stream>>>(Wv, Wvp, 256, 128);
  k_pack<<<(128 * 384 + T - 1) / T, b256, 0, stream>>>(Wi, Wip, 128, 384);
  k_pack<<<(128 * 384 + T - 1) / T, b256, 0, stream>>>(Wh, Whp, 128, 384);
  k_pack<<<(128 * 128 + T - 1) / T, b256, 0, stream>>>(Wp1, Wp1p, 128, 128);

  // h0 tile + init assignment (uniform logits -> pure gumbel argmax) + best0
  k_h0<<<2048, b256, 0, stream>>>(h_init, h_val, hv);
  k_sample<<<(NVAR + T - 1) / T, b256, 0, stream>>>(nullptr, gumbel, assign, dout, -1);
  k_unsat<<<(NCST + T - 1) / T, b256, 0, stream>>>(assign, val_idx, sat_acc);
  k_finalize<<<1, 1, 0, stream>>>(sat_acc, best, -1, dout);

  for (int t = 0; t < NSTEPS; ++t) {
    // val2cst
    k_edge_mlp1<<<ET, b256, 0, stream>>>(hv, assign, val_idx, W1p, w1a, b1, sb);
    k_edge_mlp2<<<ET, b256, 0, stream>>>(sb, W2p, r_cst);
    k_f32_to_bf16<<<2048, b256, 0, stream>>>(r_cst, rc, (size_t)NCST * H);
    // x_val
    k_gemm<<<VT, b256, 0, stream>>>(hv, 128, Wxp, 128, bx, xy, 1);
    // cst2val (scatter-add into y_val)
    k_zero_f32<<<2048, b256, 0, stream>>>(y_val, (size_t)NVAL * H);
    k_cst2val<<<ET, b256, 0, stream>>>(rc, xy, val_idx, Wcp, bc, y_val);
    // val2val
    k_ymean<<<(NVAR * H + T - 1) / T, b256, 0, stream>>>(y_val, ym);
    k_f32_to_bf16<<<2048, b256, 0, stream>>>(y_val, xy, (size_t)NVAL * H); // reuse xy as yv
    k_val2val<<<VT, b256, 0, stream>>>(xy, ym, Wvp, bv, sb);               // sb = z
    // GRU
    k_gemm<<<VT, b256, 0, stream>>>(sb, 128, Wip, 384, bi, gi, 0);
    k_gemm<<<VT, b256, 0, stream>>>(hv, 128, Whp, 384, bh, gh, 0);
    k_gru<<<2048, b256, 0, stream>>>(gi, gh, h_val, hv);
    // policy
    k_gemm<<<VT, b256, 0, stream>>>(hv, 128, Wp1p, 128, bp1, sb, 1);
    k_logits<<<(NVAL + T - 1) / T, b256, 0, stream>>>(sb, Wp2, logits);
    // sample + metrics
    k_sample<<<(NVAR + T - 1) / T, b256, 0, stream>>>(logits, gumbel + (size_t)(t + 1) * NVAL,
                                                      assign, dout, t);
    k_unsat<<<(NCST + T - 1) / T, b256, 0, stream>>>(assign, val_idx, sat_acc);
    k_finalize<<<1, 1, 0, stream>>>(sat_acc, best, t, dout);
  }
}
